// CustomLSTM_42176578847089
// MI455X (gfx1250) — matrix operand order repaired and checked by reading, not compiled, not run on hardware
//
#include <hip/hip_runtime.h>

// ---------------------------------------------------------------------------
// LSTM on MI455X (gfx1250), bf16 WMMA (v_wmma_f32_16x16x32_bf16).
//
//  * pack_x: one-time fp32->bf16 convert of x, so the recurrence hot loop is
//    pure b128 loads + WMMA.
//  * pack_gate_weights / pack_fc_weights: fp32 -> bf16 in exact per-lane WMMA
//    B-fragment layout.
//  * lstm_persistent: 64 persistent WGs x 256 threads. WG w owns hidden cols
//    [16w,16w+16) for all four gates; 196,608 B weight slice staged into LDS
//    once as [kt][gate][lane] so each K-tile's 4 gate B-tiles are immediate
//    ds_load_b128 offsets. Waves = (mt, K-quarter): each A fragment is loaded
//    from L2 exactly once and feeds 4 WMMAs (one per gate). The per-wave K
//    ranges are homogeneous (4 x-tiles + 8 h-tiles) with constant trip counts
//    so the compiler fully unrolls and software-pipelines the loads.
//    Partials reduced via LDS; cell state lives in LDS all 512 steps;
//    device-scope atomic grid barrier between time steps.
//  * fc_gemm: [16384x1024]@[1024x512] bf16 WMMA GEMM over the hs history.
// ---------------------------------------------------------------------------

typedef __attribute__((ext_vector_type(16))) __bf16 v16bf;
typedef __attribute__((ext_vector_type(8)))  float  v8f;

#define NB     32
#define SEQ    512
#define DIN    512
#define DH     1024
#define DOUT   512
#define NGW    64      // persistent workgroups (one per 16 hidden columns)
#define KT_ALL 48      // 1536 / 32 K-tiles
#define KT_X   16      // K-tiles covering the x region (512/32)

// ws layout (bytes)
#define WS_PACKW   0            // 4*1536*1024 bf16 = 12,582,912
#define WS_PACKFC  12582912     // 1024*512   bf16 =  1,048,576
#define WS_XB      13631488     // 32*512*512 bf16 = 16,777,216 (slot 33,554,432)
#define WS_HS      47185920     // 32*512*1024 bf16 = 33,554,432
#define WS_CNT     80740352     // barrier counter (4 B)

static __device__ __forceinline__ unsigned short f2bf_bits(float f) {
    unsigned u = __builtin_bit_cast(unsigned, f);
    unsigned r = u + 0x7FFFu + ((u >> 16) & 1u);      // round-to-nearest-even
    return (unsigned short)(r >> 16);
}
static __device__ __forceinline__ float sigmoid_f(float x) {
    return 1.0f / (1.0f + __expf(-x));
}
static __device__ __forceinline__ float tanh_f(float x) {
    float e = __expf(2.0f * x);
    return 1.0f - 2.0f / (e + 1.0f);
}

union FragBF {
    uint4 q[2];
    v16bf v;
};

// One A fragment vs the 4 gate B-tiles of K-tile kt (B from LDS).
static __device__ __forceinline__ void wmma_quad(
    const __bf16* __restrict__ bWlds, unsigned kt, unsigned lane,
    const FragBF& a, v8f& acc0, v8f& acc1, v8f& acc2, v8f& acc3)
{
    const uint4* pb = (const uint4*)(bWlds + (size_t)kt * 2048u + lane * 16u);
    FragBF b0, b1, b2, b3;
    b0.q[0] = pb[0];   b0.q[1] = pb[1];
    b1.q[0] = pb[64];  b1.q[1] = pb[65];
    b2.q[0] = pb[128]; b2.q[1] = pb[129];
    b3.q[0] = pb[192]; b3.q[1] = pb[193];
    acc0 = __builtin_amdgcn_wmma_f32_16x16x32_bf16(
               false, a.v, false, b0.v, (short)0, acc0, false, false);
    acc1 = __builtin_amdgcn_wmma_f32_16x16x32_bf16(
               false, a.v, false, b1.v, (short)0, acc1, false, false);
    acc2 = __builtin_amdgcn_wmma_f32_16x16x32_bf16(
               false, a.v, false, b2.v, (short)0, acc2, false, false);
    acc3 = __builtin_amdgcn_wmma_f32_16x16x32_bf16(
               false, a.v, false, b3.v, (short)0, acc3, false, false);
}

// --------------------------- input conversion -------------------------------
__global__ void pack_x(const float* __restrict__ x, __bf16* __restrict__ xb) {
    unsigned idx = blockIdx.x * 256u + threadIdx.x;   // 0 .. 16,777,215
    xb[idx] = __builtin_bit_cast(__bf16, f2bf_bits(x[idx]));
}

// --------------------------- weight packing ---------------------------------
// dst[((g*64+nt)*48+kt)*512 + lane*16 + e] = W_g[k][n]
//   n = nt*16 + (lane&15),  k = kt*32 + (lane>>4)*16 + e   (contiguous-k B-fragment, matches A loads)
__global__ void pack_gate_weights(const float* __restrict__ Wf,
                                  const float* __restrict__ Wi,
                                  const float* __restrict__ Wg,
                                  const float* __restrict__ Wo,
                                  __bf16* __restrict__ dst) {
    unsigned idx = blockIdx.x * 256u + threadIdx.x;     // 0 .. 6,291,455
    const unsigned perGate = 64u * 48u * 512u;
    unsigned g   = idx / perGate;
    unsigned r   = idx % perGate;
    unsigned nt  = r / (48u * 512u);
    unsigned r2  = r % (48u * 512u);
    unsigned kt  = r2 / 512u;
    unsigned r3  = r2 % 512u;
    unsigned ln  = r3 >> 4;
    unsigned e   = r3 & 15u;
    unsigned n   = nt * 16u + (ln & 15u);
    unsigned k   = kt * 32u + (ln >> 4) * 16u + e;
    const float* W = (g == 0) ? Wf : (g == 1) ? Wi : (g == 2) ? Wg : Wo;
    dst[idx] = __builtin_bit_cast(__bf16, f2bf_bits(W[(size_t)k * DH + n]));
}

__global__ void pack_fc_weights(const float* __restrict__ Wfc,
                                __bf16* __restrict__ dst) {
    unsigned idx = blockIdx.x * 256u + threadIdx.x;     // 0 .. 524,287
    unsigned nt  = idx / (32u * 512u);
    unsigned r2  = idx % (32u * 512u);
    unsigned kt  = r2 / 512u;
    unsigned r3  = r2 % 512u;
    unsigned ln  = r3 >> 4;
    unsigned e   = r3 & 15u;
    unsigned n   = nt * 16u + (ln & 15u);
    unsigned k   = kt * 32u + (ln >> 4) * 16u + e;
    dst[idx] = __builtin_bit_cast(__bf16, f2bf_bits(Wfc[(size_t)k * DOUT + n]));
}

__global__ void init_barrier(unsigned* cnt) { *cnt = 0u; }

// --------------------------- persistent LSTM --------------------------------
__global__ void __launch_bounds__(256, 1) lstm_persistent(
    const __bf16* __restrict__ xb,        // [32][512][512] bf16
    const __bf16* __restrict__ packedW,   // [4][64][48][32][16]
    const float* __restrict__ bfp, const float* __restrict__ bip,
    const float* __restrict__ bgp, const float* __restrict__ bop,
    __bf16* __restrict__ hs,              // [32][512][1024] bf16
    unsigned* __restrict__ barrier_cnt)
{
    // LDS weight layout: [kt(48)][gate(4)][lane(32)][16 bf16]  (196,608 B)
    __shared__ __attribute__((aligned(16))) __bf16 bWlds[KT_ALL * 4 * 512];
    __shared__ float gateT[4][4][2][16][16];   // [ks][gate][mt] partials, 32 KB
    __shared__ float cLDS[NB][16];             // 2 KB

    const unsigned tid    = threadIdx.x;
    const unsigned w      = blockIdx.x;     // hidden-column tile 0..63
    const unsigned waveId = tid >> 5;
    const unsigned lane   = tid & 31u;
    const unsigned ks     = waveId & 3u;    // K-quarter
    const unsigned mt     = waveId >> 2;    // 0/1 -> batch rows 0-15 / 16-31

    // ---- stage this WG's B-fragments (all 4 gates) into LDS, once ----
    {
        const uint4* srcW = (const uint4*)packedW;  // 8 bf16 per uint4
        uint4*       dstW = (uint4*)bWlds;
        for (unsigned i = tid; i < 12288u; i += 256u) {   // 48*4*512/8
            unsigned kt   = i >> 8;          // 256 uint4 per K-tile
            unsigned rem  = i & 255u;
            unsigned g    = rem >> 6;        // 64 uint4 per gate tile
            unsigned r    = rem & 63u;
            dstW[i] = srcW[(size_t)(g * 64u + w) * 3072u + kt * 64u + r];
        }
    }
    // zero cell state (512 floats, 2 per thread)
    cLDS[tid >> 4][tid & 15u] = 0.0f;
    cLDS[(tid + 256u) >> 4][(tid + 256u) & 15u] = 0.0f;
    __syncthreads();

    const unsigned mrow  = mt * 16u + (lane & 15u);   // batch index for A rows
    const unsigned kbase = (lane >> 4) * 16u;         // contiguous-k A-fragment half
    // per-thread biases for elementwise phase (n = tid & 15 in both halves)
    const float bias_f = bfp[w * 16u + (tid & 15u)];
    const float bias_i = bip[w * 16u + (tid & 15u)];
    const float bias_g = bgp[w * 16u + (tid & 15u)];
    const float bias_o = bop[w * 16u + (tid & 15u)];

    for (unsigned t = 0; t < SEQ; ++t) {
        v8f acc0 = {}, acc1 = {}, acc2 = {}, acc3 = {};
        const __bf16* xrow = xb + ((size_t)mrow * SEQ + t) * DIN;
        const __bf16* hrow = hs + ((size_t)mrow * SEQ + (t - 1)) * DH;

        // ---- x region: 4 K-tiles per wave, constant trip count ----
        #pragma unroll
        for (unsigned j = 0; j < 4u; ++j) {
            const unsigned kt = ks * 4u + j;
            FragBF a;
            const __bf16* p0 = xrow + kt * 32u + kbase;
            a.q[0] = *(const uint4*)(p0);
            a.q[1] = *(const uint4*)(p0 + 8);
            wmma_quad(bWlds, kt, lane, a, acc0, acc1, acc2, acc3);
        }
        // ---- h region: 8 K-tiles per wave, constant trip count (h_{-1}=0) --
        if (t > 0) {
            #pragma unroll
            for (unsigned j = 0; j < 8u; ++j) {
                const unsigned kt = KT_X + ks * 8u + j;
                FragBF a;
                const __bf16* p0 = hrow + (kt * 32u - DIN) + kbase;
                a.q[0] = *(const uint4*)(p0);
                a.q[1] = *(const uint4*)(p0 + 8);
                wmma_quad(bWlds, kt, lane, a, acc0, acc1, acc2, acc3);
            }
        }

        // ---- store partials; C/D layout: VGPR r, lanes0-15 M=r, 16-31 M=8+r
        #pragma unroll
        for (unsigned r = 0; r < 8; ++r) {
            unsigned row = (lane >> 4) * 8u + r;
            gateT[ks][0][mt][row][lane & 15u] = acc0[r];
            gateT[ks][1][mt][row][lane & 15u] = acc1[r];
            gateT[ks][2][mt][row][lane & 15u] = acc2[r];
            gateT[ks][3][mt][row][lane & 15u] = acc3[r];
        }
        __syncthreads();

        // ---- elementwise cell update: 512 elems, 2 per thread ----
        #pragma unroll
        for (unsigned ii = 0; ii < 2; ++ii) {
            unsigned e   = tid + ii * 256u;
            unsigned m   = e >> 4, n = e & 15u;
            unsigned mtt = m >> 4, rr = m & 15u;
            float pf = bias_f, pi = bias_i, pg = bias_g, po = bias_o;
            #pragma unroll
            for (unsigned q = 0; q < 4; ++q) {
                pf += gateT[q][0][mtt][rr][n];
                pi += gateT[q][1][mtt][rr][n];
                pg += gateT[q][2][mtt][rr][n];
                po += gateT[q][3][mtt][rr][n];
            }
            float fv = sigmoid_f(pf);
            float iv = sigmoid_f(pi);
            float gv = tanh_f(pg);
            float ov = sigmoid_f(po);
            float c  = fv * cLDS[m][n] + iv * gv;
            cLDS[m][n] = c;
            float h  = ov * tanh_f(c);
            hs[((size_t)m * SEQ + t) * DH + w * 16u + n] =
                __builtin_bit_cast(__bf16, f2bf_bits(h));
        }

        // ---- device-scope grid barrier between time steps ----
        __threadfence();
        __syncthreads();
        if (tid == 0) {
            __hip_atomic_fetch_add(barrier_cnt, 1u, __ATOMIC_RELEASE,
                                   __HIP_MEMORY_SCOPE_AGENT);
            const unsigned target = (t + 1u) * NGW;
            while (__hip_atomic_load(barrier_cnt, __ATOMIC_ACQUIRE,
                                     __HIP_MEMORY_SCOPE_AGENT) < target) {
                __builtin_amdgcn_s_sleep(2);
            }
        }
        __syncthreads();
        __threadfence();
    }
}

// --------------------------- final projection -------------------------------
// out[32*512, 512] = hs[32*512, 1024] @ W_fc + b_fc   (bf16 WMMA, f32 out)
__global__ void __launch_bounds__(256) fc_gemm(
    const __bf16* __restrict__ hs,
    const __bf16* __restrict__ packedWfc,   // [32][32][32][16]
    const float* __restrict__ bfc,
    float* __restrict__ out)
{
    const unsigned tid    = threadIdx.x;
    const unsigned waveId = tid >> 5;
    const unsigned lane   = tid & 31u;
    const unsigned tile   = blockIdx.x * 8u + waveId;   // 0..32767
    const unsigned tm     = tile >> 5;                  // 0..1023 (M tiles)
    const unsigned tn     = tile & 31u;                 // 0..31   (N tiles)
    const unsigned mrow   = tm * 16u + (lane & 15u);
    const unsigned kbase  = (lane >> 4) * 16u;
    const __bf16* wb   = packedWfc + (size_t)tn * 32u * 512u;
    const __bf16* arow = hs + (size_t)mrow * DH;

    v8f acc = {};
    for (unsigned kt = 0; kt < 32u; ++kt) {
        FragBF a, b;
        const __bf16* p0 = arow + kt * 32u + kbase;
        a.q[0] = *(const uint4*)(p0);
        a.q[1] = *(const uint4*)(p0 + 8);
        const uint4* pb = (const uint4*)(wb + ((size_t)kt * 32u + lane) * 16u);
        b.q[0] = pb[0]; b.q[1] = pb[1];
        acc = __builtin_amdgcn_wmma_f32_16x16x32_bf16(
                  false, a.v, false, b.v, (short)0, acc, false, false);
    }
    const float biasv = bfc[tn * 16u + (lane & 15u)];
    #pragma unroll
    for (unsigned r = 0; r < 8; ++r) {
        unsigned orow = tm * 16u + (lane >> 4) * 8u + r;
        out[(size_t)orow * DOUT + tn * 16u + (lane & 15u)] = acc[r] + biasv;
    }
}

// ----------------------------------------------------------------------------
extern "C" void kernel_launch(void* const* d_in, const int* in_sizes, int n_in,
                              void* d_out, int out_size, void* d_ws, size_t ws_size,
                              hipStream_t stream) {
    (void)in_sizes; (void)n_in; (void)out_size; (void)ws_size;
    const float* x    = (const float*)d_in[0];
    const float* W_f  = (const float*)d_in[1];
    const float* b_f  = (const float*)d_in[2];
    const float* W_i  = (const float*)d_in[3];
    const float* b_i  = (const float*)d_in[4];
    const float* W_g  = (const float*)d_in[5];
    const float* b_g  = (const float*)d_in[6];
    const float* W_o  = (const float*)d_in[7];
    const float* b_o  = (const float*)d_in[8];
    const float* W_fc = (const float*)d_in[9];
    const float* b_fc = (const float*)d_in[10];

    char* ws = (char*)d_ws;                       // needs ~80.8 MB
    __bf16*   packedW   = (__bf16*)(ws + WS_PACKW);
    __bf16*   packedWfc = (__bf16*)(ws + WS_PACKFC);
    __bf16*   xb        = (__bf16*)(ws + WS_XB);
    __bf16*   hs        = (__bf16*)(ws + WS_HS);
    unsigned* cnt       = (unsigned*)(ws + WS_CNT);

    init_barrier<<<1, 1, 0, stream>>>(cnt);
    pack_x<<<65536, 256, 0, stream>>>(x, xb);
    pack_gate_weights<<<24576, 256, 0, stream>>>(W_f, W_i, W_g, W_o, packedW);
    pack_fc_weights<<<2048, 256, 0, stream>>>(W_fc, packedWfc);
    lstm_persistent<<<NGW, 256, 0, stream>>>(xb, packedW, b_f, b_i, b_g, b_o,
                                             hs, cnt);
    fc_gemm<<<4096, 256, 0, stream>>>(hs, packedWfc, b_fc, (float*)d_out);
}
